// DeformConv_49366354100570
// MI455X (gfx1250) — matrix operand order repaired and checked by reading, not compiled, not run on hardware
//
#include <hip/hip_runtime.h>

// ---------------------------------------------------------------------------
// Fused modulated deformable conv (DCNv2) + BN + ReLU for gfx1250 (MI455X)
// bf16 WMMA main path, f32 accumulate, cols generated in-LDS (never hit HBM).
// Weight tiles move via global_load_async_to_lds_b128 with a double-buffered
// software pipeline: chunk i+1's DMA overlaps chunk i's sampling AND WMMA
// (s_wait_asynccnt 4 retires only the older batch).
// ---------------------------------------------------------------------------

typedef __attribute__((ext_vector_type(16))) __bf16 v16bf;
typedef __attribute__((ext_vector_type(8)))  float  v8f;

#define BATCH 2
#define CIN   256
#define COUT  256
#define HH    96
#define WW    96
#define HWSZ  (HH*WW)            // 9216
#define KKC   9
#define CKDIM (CIN*KKC)          // 2304
#define NITER (CKDIM/32)         // 72 K-chunks
#define NPIX  (BATCH*HWSZ)       // 18432
#define NOFF  27
#define NOFFP 32                 // padded offset-conv output rows

#define NX       (BATCH*CIN*HWSZ)   // 4,718,592
#define NWM      (COUT*CKDIM)       // 589,824
#define NWOFF    (NOFF*CKDIM)       // 62,208
#define NWOFFP   (NOFFP*CKDIM)      // 73,728 (zero-padded rows 27..31)
#define NOFFMAP  (BATCH*KKC*HWSZ)   // 165,888 per array

// workspace layout (bytes)
static constexpr size_t XB_OFF    = 0;
static constexpr size_t WM_OFF    = XB_OFF   + (size_t)NX     * 2;
static constexpr size_t WOFF_OFF  = WM_OFF   + (size_t)NWM    * 2;
static constexpr size_t OFFY_OFF  = WOFF_OFF + (size_t)NWOFFP * 2;
static constexpr size_t OFFX_OFF  = OFFY_OFF + (size_t)NOFFMAP * 4;
static constexpr size_t MASK_OFF  = OFFX_OFF + (size_t)NOFFMAP * 4;
static constexpr size_t SCALE_OFF = MASK_OFF + (size_t)NOFFMAP * 4;
static constexpr size_t SHIFT_OFF = SCALE_OFF + (size_t)COUT * 4;

// --- CDNA5 fragment swizzles (ISA 7.12.2, 16-bit operands) ------------------
// A 16x32 bf16: lane = (m&15) + 16*((k>>3)&1) ; elem = (k&7) | ((k>>4)<<3)
//   -> each (m, k-octet) is 16 contiguous bytes in LDS AND in row-major global
// B 32x16 bf16: lane = n + 16*((k>>3)&1)      ; elem = (k&7) | ((k>>4)<<3)
// D 16x16 f32 : lane = n + 16*(m>>3)          ; vgpr = m & 7
__device__ __forceinline__ int b_swz(int n, int k) {
  return ((n >> 4) * 32 + ((n & 15) + (((k >> 3) & 1) << 4))) * 16 + ((k & 7) | ((k >> 4) << 3));
}
// LDS byte offset of an (m, octet) A-fragment block (16B, b128-contiguous)
__device__ __forceinline__ unsigned a_blk_off(int m, int oct) {
  return (unsigned)((((m >> 4) * 32 + ((m & 15) + ((oct & 1) << 4))) * 16
                     + ((oct >> 1) << 3)) * 2);
}

// async 16B global->LDS copy (CDNA5 async engine, tracked by ASYNCcnt).
// Low 32 bits of a generic LDS pointer are the LDS byte offset (ISA 10.2).
__device__ __forceinline__ void async_copy_b128(unsigned lds_byte_off,
                                                const void* gptr) {
  asm volatile("global_load_async_to_lds_b128 %0, %1, off"
               :: "v"(lds_byte_off), "v"(gptr) : "memory");
}
__device__ __forceinline__ void wait_async0() {   // all asyncs retired
  asm volatile("s_wait_asynccnt 0x0" ::: "memory");
}
__device__ __forceinline__ void wait_async4() {   // allow 4 newest in flight
  asm volatile("s_wait_asynccnt 0x4" ::: "memory");
}

// ---------------------------------------------------------------------------
// Kernel 1: pack fp32 -> bf16 (w_off zero-padded to 32 rows) + BN scale/shift
// ---------------------------------------------------------------------------
__global__ __launch_bounds__(256) void pack_kernel(
    const float* __restrict__ x, const float* __restrict__ w,
    const float* __restrict__ w_off, const float* __restrict__ bias,
    const float* __restrict__ gamma, const float* __restrict__ beta,
    const float* __restrict__ rmean, const float* __restrict__ rvar,
    __bf16* __restrict__ xb, __bf16* __restrict__ wmb,
    __bf16* __restrict__ wob, float* __restrict__ scale,
    float* __restrict__ shift) {
  int i = blockIdx.x * 256 + threadIdx.x;
  if (i < NX)     xb[i]  = (__bf16)x[i];
  if (i < NWM)    wmb[i] = (__bf16)w[i];
  if (i < NWOFFP) wob[i] = (i < NWOFF) ? (__bf16)w_off[i] : (__bf16)0.0f;
  if (i < COUT) {
    float inv = gamma[i] * rsqrtf(rvar[i] + 1e-5f);
    scale[i] = inv;
    shift[i] = beta[i] + (bias[i] - rmean[i]) * inv;  // fold conv bias into BN
  }
}

// ---------------------------------------------------------------------------
// Kernel 2: offset conv (27->pad32 out ch) as implicit GEMM via bf16 WMMA
// ---------------------------------------------------------------------------
__global__ __launch_bounds__(256) void offset_conv_kernel(
    const __bf16* __restrict__ xb, const __bf16* __restrict__ wob,
    const float* __restrict__ b_off, float* __restrict__ offy,
    float* __restrict__ offx, float* __restrict__ maskv) {
  __shared__ __align__(32) __bf16 ldsA[2 * 512];   // 32 x 32 (2 M-tiles)
  __shared__ __align__(32) __bf16 ldsB[8 * 512];   // 32 x 128 (8 N-tiles)
  const int t = threadIdx.x, lane = t & 31, wid = t >> 5;
  const int pixBase = blockIdx.x * 128;            // 9216 % 128 == 0
  const int bIdx = pixBase / HWSZ;
  const unsigned ldsA_base = (unsigned)(uintptr_t)&ldsA[0];

  v8f acc0 = {}, acc1 = {};
  for (int ck0 = 0; ck0 < CKDIM; ck0 += 32) {
    // async-stage A: 32x32 weight tile = 128 x 16B blocks (threads 0..127)
    if (t < 128) {
      int m = t >> 2, oct = t & 3;
      async_copy_b128(ldsA_base + a_blk_off(m, oct),
                      wob + (size_t)m * CKDIM + ck0 + oct * 8);
    }
    // stage B: 32 x 128 im2col tile (branchless zero padding), overlaps DMA
    #pragma unroll
    for (int j = 0; j < 16; ++j) {
      int idx = j * 256 + t;                 // idx = kloc*128 + n
      int kloc = idx >> 7, n = idx & 127;
      int ck = ck0 + kloc;
      int c = ck / KKC, kk = ck - c * KKC;
      int hw = pixBase + n - bIdx * HWSZ;
      int h = hw / WW, wcol = hw - h * WW;
      int yy = h + kk / 3 - 1;
      int xx = wcol + (kk - (kk / 3) * 3) - 1;
      bool ok = ((unsigned)yy < HH) & ((unsigned)xx < WW);
      int yc = ok ? yy : 0, xc = ok ? xx : 0;
      __bf16 v = xb[((bIdx * CIN + c) * HH + yc) * WW + xc];  // unconditional
      ldsB[b_swz(n, kloc)] = ok ? v : (__bf16)0.0f;
    }
    wait_async0();
    __syncthreads();
    v16bf a0 = *(const v16bf*)&ldsA[(0 * 32 + lane) * 16];
    v16bf a1 = *(const v16bf*)&ldsA[(1 * 32 + lane) * 16];
    v16bf bb = *(const v16bf*)&ldsB[(wid * 32 + lane) * 16];  // ntile = wid
    acc0 = __builtin_amdgcn_wmma_f32_16x16x32_bf16(false, a0, false, bb,
                                                   (short)0, acc0, false, false);
    acc1 = __builtin_amdgcn_wmma_f32_16x16x32_bf16(false, a1, false, bb,
                                                   (short)0, acc1, false, false);
    __syncthreads();
  }

  // epilogue: D layout -> channel split (off_y/off_x interleaved, sigmoid mask)
  const int n = lane & 15;
  const int mhi = (lane >> 4) << 3;
  const int hw = pixBase + wid * 16 + n - bIdx * HWSZ;
  #pragma unroll
  for (int r = 0; r < 8; ++r) {
    #pragma unroll
    for (int half = 0; half < 2; ++half) {
      int ch = half * 16 + mhi + r;
      float v = (half ? acc1[r] : acc0[r]);
      if (ch < 18) {
        v += b_off[ch];
        int kk = ch >> 1;
        float* dst = (ch & 1) ? offx : offy;   // om[:, :18] -> (KK, 2) pairs
        dst[(bIdx * KKC + kk) * HWSZ + hw] = v;
      } else if (ch < NOFF) {
        v += b_off[ch];
        maskv[(bIdx * KKC + (ch - 18)) * HWSZ + hw] =
            __builtin_amdgcn_rcpf(1.0f + __expf(-v));   // v_rcp_f32 sigmoid
      }
    }
  }
}

// ---------------------------------------------------------------------------
// Kernel 3: fused deformable GEMM.  Block tile: 256 (all O) x 64 pixels.
// 8 waves, 4x2 grid, 8 WMMA D-tiles per wave.  Double-buffered async weight
// pipeline; cols tile generated in-LDS by branchless modulated bilinear
// sampling from L2-resident bf16 x.  Epilogue fuses bias+BN+ReLU.
// ---------------------------------------------------------------------------
__global__ __launch_bounds__(256) void deform_gemm_kernel(
    const __bf16* __restrict__ xb, const __bf16* __restrict__ wmb,
    const float* __restrict__ offy, const float* __restrict__ offx,
    const float* __restrict__ maskv, const float* __restrict__ scale,
    const float* __restrict__ shift, float* __restrict__ out) {
  __shared__ __align__(32) __bf16 ldsA[2][16 * 512]; // 2 x (256x32) = 32 KB
  __shared__ __align__(32) __bf16 ldsB[4 * 512];     // 32 x 64 cols  (4 KB)
  __shared__ float sOffY[KKC][64], sOffX[KKC][64], sMask[KKC][64];

  const int t = threadIdx.x, lane = t & 31, wid = t >> 5;
  const int waveM = wid & 3, waveN = wid >> 2;     // 4 x 2 wave grid
  const int pixBase = blockIdx.x * 64;             // 9216 % 64 == 0
  const int bIdx = pixBase / HWSZ;
  const unsigned ldsA_base = (unsigned)(uintptr_t)&ldsA[0][0];

  // per-thread constants for the async A staging (4 x 16B blocks / thread)
  const int am0 = t >> 2, aoct = t & 3;            // + j*64 rows per block

  // issue one 256x32 weight tile into buffer `buf` (4 asyncs per thread)
  auto issueA = [&](int ck0, int buf) {
    #pragma unroll
    for (int j = 0; j < 4; ++j) {
      int m = am0 + j * 64;
      async_copy_b128(ldsA_base + (unsigned)buf * 16384u + a_blk_off(m, aoct),
                      wmb + (size_t)m * CKDIM + ck0 + aoct * 8);
    }
  };

  // stage per-pixel deform state once per block
  for (int j = t; j < KKC * 64; j += 256) {
    int kk = j >> 6, n = j & 63;
    int g = (bIdx * KKC + kk) * HWSZ + (pixBase + n - bIdx * HWSZ);
    sOffY[kk][n] = offy[g];
    sOffX[kk][n] = offx[g];
    sMask[kk][n] = maskv[g];
  }

  v8f acc[4][2] = {};
  const __bf16* __restrict__ xbatch = xb + (size_t)bIdx * CIN * HWSZ;

  issueA(0, 0);                                    // prologue: prefetch chunk 0
  __syncthreads();                                 // sOff* visible

  for (int it = 0; it < NITER; ++it) {
    const int ck0 = it * 32;
    const int buf = it & 1;
    const bool more = (it + 1 < NITER);
    if (more) issueA(ck0 + 32, buf ^ 1);           // DMA for next chunk flies
                                                   // through sampling AND WMMA
    // ---- stage B: 32 x 64 cols via branchless bilinear sampling ----
    {
      int kloc = t >> 3;                     // each thread: 1 ck x 8 pixels
      int ck = ck0 + kloc;
      int c = ck / KKC, kk = ck - c * KKC;
      float ky = (float)(kk / 3 - 1);
      float kx = (float)(kk - (kk / 3) * 3 - 1);
      const __bf16* __restrict__ xc = xbatch + (size_t)c * HWSZ;
      #pragma unroll
      for (int j = 0; j < 8; ++j) {
        int n = ((t & 7) << 3) + j;
        int hw = pixBase + n - bIdx * HWSZ;
        int h = hw / WW, wcol = hw - h * WW;
        float py = (float)h + ky + sOffY[kk][n];
        float px = (float)wcol + kx + sOffX[kk][n];
        float y0f = floorf(py), x0f = floorf(px);
        int y0 = (int)y0f, x0 = (int)x0f;
        float wy1 = py - y0f, wx1 = px - x0f;
        float wy0 = 1.0f - wy1, wx0 = 1.0f - wx1;
        // branchless 4-tap gather: clamp index, zero the weight when OOB
        int y0c = min(max(y0, 0), HH - 1), y1c = min(max(y0 + 1, 0), HH - 1);
        int x0c = min(max(x0, 0), WW - 1), x1c = min(max(x0 + 1, 0), WW - 1);
        float vy0 = ((unsigned)y0 < HH) ? 1.0f : 0.0f;
        float vy1 = ((unsigned)(y0 + 1) < HH) ? 1.0f : 0.0f;
        float vx0 = ((unsigned)x0 < WW) ? 1.0f : 0.0f;
        float vx1 = ((unsigned)(x0 + 1) < WW) ? 1.0f : 0.0f;
        float s = (wy0 * vy0) * (wx0 * vx0) * (float)xc[y0c * WW + x0c]
                + (wy0 * vy0) * (wx1 * vx1) * (float)xc[y0c * WW + x1c]
                + (wy1 * vy1) * (wx0 * vx0) * (float)xc[y1c * WW + x0c]
                + (wy1 * vy1) * (wx1 * vx1) * (float)xc[y1c * WW + x1c];
        ldsB[b_swz(n, kloc)] = (__bf16)(s * sMask[kk][n]);
      }
    }
    // retire only the OLDER async batch (4 newest stay in flight)
    if (more) wait_async4(); else wait_async0();
    __syncthreads();

    // ---- WMMA: 4 A-frags x 2 B-frags per wave (next DMA still in flight) ----
    v16bf af[4];
    #pragma unroll
    for (int i = 0; i < 4; ++i)
      af[i] = *(const v16bf*)&ldsA[buf][((waveM * 4 + i) * 32 + lane) * 16];
    v16bf bf0 = *(const v16bf*)&ldsB[((waveN * 2 + 0) * 32 + lane) * 16];
    v16bf bf1 = *(const v16bf*)&ldsB[((waveN * 2 + 1) * 32 + lane) * 16];
    #pragma unroll
    for (int i = 0; i < 4; ++i) {
      acc[i][0] = __builtin_amdgcn_wmma_f32_16x16x32_bf16(
          false, af[i], false, bf0, (short)0, acc[i][0], false, false);
      acc[i][1] = __builtin_amdgcn_wmma_f32_16x16x32_bf16(
          false, af[i], false, bf1, (short)0, acc[i][1], false, false);
    }
    __syncthreads();
  }

  // ---- epilogue: bias+BN folded scale/shift, ReLU, NCHW store ----
  const int n = lane & 15;
  const int mhi = (lane >> 4) << 3;
  #pragma unroll
  for (int i = 0; i < 4; ++i) {
    #pragma unroll
    for (int jt = 0; jt < 2; ++jt) {
      int hw = pixBase + (waveN * 2 + jt) * 16 + n - bIdx * HWSZ;
      #pragma unroll
      for (int r = 0; r < 8; ++r) {
        int o = (waveM * 4 + i) * 16 + mhi + r;
        float v = acc[i][jt][r] * scale[o] + shift[o];
        out[((size_t)bIdx * COUT + o) * HWSZ + hw] = fmaxf(v, 0.0f);
      }
    }
  }
}

// ---------------------------------------------------------------------------
extern "C" void kernel_launch(void* const* d_in, const int* in_sizes, int n_in,
                              void* d_out, int out_size, void* d_ws,
                              size_t ws_size, hipStream_t stream) {
  const float* x      = (const float*)d_in[0];
  const float* w_off  = (const float*)d_in[1];
  const float* b_off  = (const float*)d_in[2];
  const float* w      = (const float*)d_in[3];
  const float* bias   = (const float*)d_in[4];
  const float* gamma  = (const float*)d_in[5];
  const float* beta   = (const float*)d_in[6];
  const float* rmean  = (const float*)d_in[7];
  const float* rvar   = (const float*)d_in[8];

  char* ws = (char*)d_ws;
  __bf16* xb   = (__bf16*)(ws + XB_OFF);
  __bf16* wmb  = (__bf16*)(ws + WM_OFF);
  __bf16* wob  = (__bf16*)(ws + WOFF_OFF);
  float* offy  = (float*)(ws + OFFY_OFF);
  float* offx  = (float*)(ws + OFFX_OFF);
  float* maskv = (float*)(ws + MASK_OFF);
  float* scl   = (float*)(ws + SCALE_OFF);
  float* shf   = (float*)(ws + SHIFT_OFF);

  pack_kernel<<<NX / 256, 256, 0, stream>>>(x, w, w_off, bias, gamma, beta,
                                            rmean, rvar, xb, wmb, wob, scl, shf);
  offset_conv_kernel<<<NPIX / 128, 256, 0, stream>>>(xb, wob, b_off, offy,
                                                     offx, maskv);
  deform_gemm_kernel<<<NPIX / 64, 256, 0, stream>>>(xb, wmb, offy, offx, maskv,
                                                    scl, shf, (float*)d_out);
}
